// GlycanGraphEncoder_41429254537575
// MI455X (gfx1250) — matrix operand order repaired and checked by reading, not compiled, not run on hardware
//
#include <hip/hip_runtime.h>
#include <math.h>

// ---------------- CDNA5 WMMA types ----------------
typedef __bf16 bf16_t;
typedef __attribute__((ext_vector_type(16))) __bf16 v16bf;
typedef __attribute__((ext_vector_type(8)))  float  v8f;

#define HID    128
#define NGAUSS 50
#define KG     64        // NGAUSS padded to WMMA K granularity
#define OUTD   256
#define METAD  11
#define NPG    128       // atoms per graph
#define WPB    4         // waves per block for the heavy kernels

union FragU { float4 f[2]; v16bf v; };

// A-matrix 16x32 bf16 fragment (per ISA 7.12.2):
// lanes 0-15 : M=lane,    elems 0..7 = K kb+0..7,  elems 8..15 = K kb+16..23
// lanes16-31 : M=lane-16, elems 0..7 = K kb+8..15, elems 8..15 = K kb+24..31
__device__ __forceinline__ v16bf ld_a_frag(const bf16_t* p, int ld, int m, int hi, int kb) {
  int k0 = kb + hi * 8;
  FragU u;
  u.f[0] = *reinterpret_cast<const float4*>(p + m * ld + k0);
  u.f[1] = *reinterpret_cast<const float4*>(p + m * ld + k0 + 16);
  return u.v;
}

// B-matrix 32x16 bf16 fragment: column n = lane%16,
// lanes 0-15: elems 0..7 = K kb+0..7, elems 8..15 = K kb+16..23;
// lanes 16-31: elems 0..7 = K kb+8..15, elems 8..15 = K kb+24..31 (matches A).
// Weights staged transposed in LDS as [n][k] so K is contiguous.
__device__ __forceinline__ v16bf ld_b_frag(const bf16_t* p, int ld, int n, int hi, int kb) {
  int k0 = kb + hi * 8;
  FragU u;
  u.f[0] = *reinterpret_cast<const float4*>(p + n * ld + k0);
  u.f[1] = *reinterpret_cast<const float4*>(p + n * ld + k0 + 16);
  return u.v;
}

__device__ __forceinline__ v8f wmma_bf16(v16bf a, v16bf b, v8f c) {
  return __builtin_amdgcn_wmma_f32_16x16x32_bf16(false, a, false, b, (short)0, c, false, false);
}

// softplus(v) - ln2, branch-free stable form: max(v,0) + log(1+exp(-|v|)) - ln2
__device__ __forceinline__ float sspf(float v) {
  return fmaxf(v, 0.f) + __logf(1.f + __expf(-fabsf(v))) - 0.69314718056f;
}

// ---------------- small helper kernels ----------------
__global__ void k_zero(float* p, int n) {
  int i = blockIdx.x * blockDim.x + threadIdx.x;
  if (i < n) p[i] = 0.f;
}

__global__ void k_embed(const int* z, const float* emb, float* h, int n) {
  int i = blockIdx.x * blockDim.x + threadIdx.x;
  if (i < n) h[i] = emb[z[i >> 7] * HID + (i & 127)];
}

// ---------------- generic node GEMM: C = f(A[M,128] @ B[128,128] + bias) ----------------
// MODE 0: store   MODE 1: ssp then store   MODE 2: Cout += result
template <int MODE, bool HAS_BIAS>
__global__ __launch_bounds__(32 * WPB)
void k_node_gemm(const float* __restrict__ A, const float* __restrict__ Bw,
                 const float* __restrict__ bias, float* __restrict__ Cout,
                 int nRows) {
  __shared__ __align__(16) bf16_t wt[HID * HID];        // B transposed [n][k], 32 KB
  __shared__ __align__(16) bf16_t at[WPB][16 * HID];    // per-wave A tile, 16 KB
  int tid = threadIdx.x;
  __builtin_prefetch(Bw, 0, 1);
  for (int i = tid; i < HID * HID; i += blockDim.x) {
    int n = i >> 7, k = i & 127;
    wt[i] = (bf16_t)Bw[k * HID + n];
  }
  __syncthreads();

  int wave = tid >> 5, lane = tid & 31, hi = lane >> 4, lm = lane & 15;
  int row0 = (blockIdx.x * WPB + wave) * 16;
  if (row0 >= nRows) return;

  bf16_t* a = at[wave];
  for (int i = lane; i < 16 * HID; i += 32)
    a[i] = (bf16_t)A[(row0 + (i >> 7)) * HID + (i & 127)];

  // A fragments are invariant across column tiles: load once
  v16bf fa[4];
#pragma unroll
  for (int kk = 0; kk < 4; ++kk) fa[kk] = ld_a_frag(a, HID, lm, hi, kk * 32);

#pragma unroll
  for (int ct = 0; ct < 8; ++ct) {
    v8f acc = {};
#pragma unroll
    for (int kk = 0; kk < 4; ++kk) {
      v16bf fb = ld_b_frag(wt, HID, ct * 16 + lm, hi, kk * 32);
      acc = wmma_bf16(fa[kk], fb, acc);
    }
    int col = ct * 16 + lm;
    float bb = HAS_BIAS ? bias[col] : 0.f;
#pragma unroll
    for (int r = 0; r < 8; ++r) {
      int row = row0 + hi * 8 + r;
      float v = acc[r] + bb;
      if (MODE == 1) v = sspf(v);
      if (MODE == 2) Cout[row * HID + col] += v;
      else           Cout[row * HID + col]  = v;
    }
  }
}

// ---------------- fused edge kernel (one 16-edge tile per wave) ----------------
// W = ssp(edge_attr @ W1 + b1) @ W2 + b2, scaled by cosine cutoff,
// then agg[dst] += x[src] * W   (fp32 atomics, L2 resident)
__global__ __launch_bounds__(32 * WPB)
void k_edge(const float* __restrict__ pos, const int* __restrict__ eidx, int E,
            const float* __restrict__ W1, const float* __restrict__ b1,
            const float* __restrict__ W2, const float* __restrict__ b2,
            const float* __restrict__ x, float* __restrict__ agg) {
  __shared__ __align__(16) bf16_t w1t[HID * KG];        // [n<128][k<64], 16 KB
  __shared__ __align__(16) bf16_t w2t[HID * HID];       // [n<128][k<128], 32 KB
  __shared__ __align__(16) bf16_t scr[WPB][16 * HID];   // per-wave scratch, 16 KB
  int tid = threadIdx.x;
  __builtin_prefetch(W1, 0, 1);
  __builtin_prefetch(W2, 0, 1);
  for (int i = tid; i < HID * KG; i += blockDim.x) {
    int n = i >> 6, k = i & 63;
    w1t[i] = (bf16_t)((k < NGAUSS) ? W1[k * HID + n] : 0.f);
  }
  for (int i = tid; i < HID * HID; i += blockDim.x) {
    int n = i >> 7, k = i & 127;
    w2t[i] = (bf16_t)W2[k * HID + n];
  }
  __syncthreads();

  int wave = tid >> 5, lane = tid & 31, hi = lane >> 4, lm = lane & 15;
  int e0 = (blockIdx.x * WPB + wave) * 16;
  if (e0 >= E) return;

  // per-lane edge data on lanes 0..15 (padded edges get C=0 -> contribute 0)
  int e = e0 + lm;
  bool valid = (e < E) && (lane < 16);
  int  si = valid ? eidx[e]     : 0;
  int  di = valid ? eidx[E + e] : 0;
  float d = 0.f, Cc = 0.f;
  if (valid) {
    float dx = pos[si * 3 + 0] - pos[di * 3 + 0];
    float dy = pos[si * 3 + 1] - pos[di * 3 + 1];
    float dz = pos[si * 3 + 2] - pos[di * 3 + 2];
    d  = sqrtf(dx * dx + dy * dy + dz * dz);
    Cc = 0.5f * (__cosf(d * 0.6283185307f) + 1.f);  // pi/CUT
  }

  // edge_attr tile [16 x 64] (gaussian smearing, zero-padded K)
  const float GDELTA = 5.0f / 49.0f;
  const float GCOEF  = -0.5f / (GDELTA * GDELTA);
  bf16_t* sc = scr[wave];
#pragma unroll
  for (int it = 0; it < 16 * KG / 32; ++it) {
    int i = it * 32 + lane;
    int row = i >> 6, k = i & 63;
    float dr = __shfl(d, row, 32);
    float t = dr - (float)k * GDELTA;
    float v = (k < NGAUSS) ? __expf(GCOEF * t * t) : 0.f;
    sc[row * KG + k] = (bf16_t)v;
  }

  // GEMM1: [16,64] @ [64,128] -> keep all 8 accumulators so scr can be reused
  v16bf fa0 = ld_a_frag(sc, KG, lm, hi, 0);
  v16bf fa1 = ld_a_frag(sc, KG, lm, hi, 32);
  v8f acc1[8];
#pragma unroll
  for (int ct = 0; ct < 8; ++ct) {
    v8f acc = {};
    acc = wmma_bf16(fa0, ld_b_frag(w1t, KG, ct * 16 + lm, hi, 0),  acc);
    acc = wmma_bf16(fa1, ld_b_frag(w1t, KG, ct * 16 + lm, hi, 32), acc);
    acc1[ct] = acc;
  }
  // intermediate = ssp(. + b1) -> scr (stride 128, overwrites attr tile)
#pragma unroll
  for (int ct = 0; ct < 8; ++ct) {
    int col = ct * 16 + lm;
    float bb = b1[col];
#pragma unroll
    for (int r = 0; r < 8; ++r)
      sc[(hi * 8 + r) * HID + col] = (bf16_t)sspf(acc1[ct][r] + bb);
  }

  // GEMM2 + cutoff scale + gather/scatter (A fragments hoisted)
  v16bf ga[4];
#pragma unroll
  for (int kk = 0; kk < 4; ++kk) ga[kk] = ld_a_frag(sc, HID, lm, hi, kk * 32);

#pragma unroll
  for (int ct = 0; ct < 8; ++ct) {
    v8f acc = {};
#pragma unroll
    for (int kk = 0; kk < 4; ++kk) {
      v16bf fb = ld_b_frag(w2t, HID, ct * 16 + lm, hi, kk * 32);
      acc = wmma_bf16(ga[kk], fb, acc);
    }
    int col = ct * 16 + lm;
    float bb = b2[col];
#pragma unroll
    for (int r = 0; r < 8; ++r) {
      int erow = hi * 8 + r;
      float Ce = __shfl(Cc, erow, 32);
      int   ss = __shfl(si, erow, 32);
      int   dd = __shfl(di, erow, 32);
      float W  = (acc[r] + bb) * Ce;
      float xv = x[ss * HID + col];
      atomicAdd(&agg[dd * HID + col], xv * W);
    }
  }
}

// ---------------- fused atomwise output MLP + per-graph sum (one block per graph) ----------------
#define OWPB 2
__global__ __launch_bounds__(32 * OWPB)
void k_output(const float* __restrict__ h,
              const float* __restrict__ w1, const float* __restrict__ b1,
              const float* __restrict__ w2, const float* __restrict__ b2,
              float* __restrict__ gsum) {
  __shared__ __align__(16) bf16_t w1t[64 * HID];        // out_w1^T [n<64][k<128], 16 KB
  __shared__ __align__(16) bf16_t w2t[OUTD * 64];       // out_w2^T [n<256][k<64], 32 KB
  __shared__ __align__(16) bf16_t at[OWPB][16 * HID];   // A tile / intermediate, 8 KB
  int g = blockIdx.x, tid = threadIdx.x;
  for (int i = tid; i < 64 * HID; i += blockDim.x) {
    int n = i >> 7, k = i & 127;
    w1t[i] = (bf16_t)w1[k * 64 + n];
  }
  for (int i = tid; i < OUTD * 64; i += blockDim.x) {
    int n = i >> 6, k = i & 63;
    w2t[i] = (bf16_t)w2[k * OUTD + n];
  }
  __syncthreads();

  int wave = tid >> 5, lane = tid & 31, hi = lane >> 4, lm = lane & 15;
  for (int rt = wave; rt < NPG / 16; rt += OWPB) {
    int row0 = g * NPG + rt * 16;
    bf16_t* a = at[wave];
    for (int i = lane; i < 16 * HID; i += 32)
      a[i] = (bf16_t)h[(row0 + (i >> 7)) * HID + (i & 127)];

    // GEMM1: [16,128]@[128,64]; A frags hoisted; keep 4 accumulators
    v16bf fa[4];
#pragma unroll
    for (int kk = 0; kk < 4; ++kk) fa[kk] = ld_a_frag(a, HID, lm, hi, kk * 32);
    v8f acc1[4];
#pragma unroll
    for (int ct = 0; ct < 4; ++ct) {
      v8f acc = {};
#pragma unroll
      for (int kk = 0; kk < 4; ++kk)
        acc = wmma_bf16(fa[kk], ld_b_frag(w1t, HID, ct * 16 + lm, hi, kk * 32), acc);
      acc1[ct] = acc;
    }
    // intermediate = ssp(.+b1) -> reuse `a` (stride 64)
#pragma unroll
    for (int ct = 0; ct < 4; ++ct) {
      int col = ct * 16 + lm;
      float bb = b1[col];
#pragma unroll
      for (int r = 0; r < 8; ++r)
        a[(hi * 8 + r) * 64 + col] = (bf16_t)sspf(acc1[ct][r] + bb);
    }
    // GEMM2: [16,64]@[64,256]; reduce rows in-register, accumulate graph sum
    v16bf gb0 = ld_a_frag(a, 64, lm, hi, 0);
    v16bf gb1 = ld_a_frag(a, 64, lm, hi, 32);
#pragma unroll
    for (int ct = 0; ct < 16; ++ct) {
      v8f acc = {};
      acc = wmma_bf16(gb0, ld_b_frag(w2t, 64, ct * 16 + lm, hi, 0),  acc);
      acc = wmma_bf16(gb1, ld_b_frag(w2t, 64, ct * 16 + lm, hi, 32), acc);
      int col = ct * 16 + lm;
      float bb = b2[col];
      float s = 0.f;
#pragma unroll
      for (int r = 0; r < 8; ++r) s += acc[r] + bb;  // 16 rows/tile * 8 tiles = 128 bias adds
      atomicAdd(&gsum[g * OUTD + col], s);
    }
  }
}

// ---------------- final projection: out = [gsum | meta] @ proj_w + proj_b ----------------
__global__ void k_proj(const float* __restrict__ gsum, const float* __restrict__ meta,
                       const float* __restrict__ pw, const float* __restrict__ pb,
                       float* __restrict__ out, int G) {
  int idx = blockIdx.x * blockDim.x + threadIdx.x;
  if (idx >= G * OUTD) return;
  int g = idx >> 8, o = idx & 255;
  float acc = pb[o];
  for (int k = 0; k < OUTD; ++k)   acc += gsum[g * OUTD + k] * pw[k * OUTD + o];
  for (int k = 0; k < METAD; ++k)  acc += meta[g * METAD + k] * pw[(OUTD + k) * OUTD + o];
  out[idx] = acc;
}

// ---------------- host orchestration ----------------
extern "C" void kernel_launch(void* const* d_in, const int* in_sizes, int n_in,
                              void* d_out, int out_size, void* d_ws, size_t ws_size,
                              hipStream_t stream) {
  (void)n_in; (void)out_size; (void)ws_size;
  const int*   z      = (const int*)  d_in[0];
  const float* pos    = (const float*)d_in[1];
  /* batch_idx d_in[2] unused: atoms are contiguous per graph */
  const float* meta   = (const float*)d_in[3];
  const int*   eidx   = (const int*)  d_in[4];
  const float* emb    = (const float*)d_in[5];
  const float* mlp_w1 = (const float*)d_in[6];
  const float* mlp_b1 = (const float*)d_in[7];
  const float* mlp_w2 = (const float*)d_in[8];
  const float* mlp_b2 = (const float*)d_in[9];
  const float* cf_w1  = (const float*)d_in[10];
  const float* cf_w2  = (const float*)d_in[11];
  const float* cf_b2  = (const float*)d_in[12];
  const float* int_w  = (const float*)d_in[13];
  const float* int_b  = (const float*)d_in[14];
  const float* out_w1 = (const float*)d_in[15];
  const float* out_b1 = (const float*)d_in[16];
  const float* out_w2 = (const float*)d_in[17];
  const float* out_b2 = (const float*)d_in[18];
  const float* proj_w = (const float*)d_in[19];
  const float* proj_b = (const float*)d_in[20];

  const int N = in_sizes[0];           // 16384
  const int G = N / NPG;               // 128
  const int E = in_sizes[4] / 2;

  float* ws   = (float*)d_ws;
  float* h    = ws;
  float* x    = ws + (size_t)N * HID;          // also reused as post-agg tmp
  float* agg  = ws + (size_t)2 * N * HID;
  float* gsum = ws + (size_t)3 * N * HID;

  const int NB = 256;
  k_embed<<<(N * HID + NB - 1) / NB, NB, 0, stream>>>(z, emb, h, N * HID);

  const int nodeBlocks = (N / 16 + WPB - 1) / WPB;
  for (int l = 0; l < 6; ++l) {
    k_zero<<<(N * HID + NB - 1) / NB, NB, 0, stream>>>(agg, N * HID);
    // x = h @ cf_w1[l]
    k_node_gemm<0, false><<<nodeBlocks, 32 * WPB, 0, stream>>>(
        h, cf_w1 + (size_t)l * HID * HID, nullptr, x, N);
    // agg[dst] += x[src] * CFConv-filter(edge)
    int nTiles = (E + 15) / 16;
    k_edge<<<(nTiles + WPB - 1) / WPB, 32 * WPB, 0, stream>>>(
        pos, eidx, E,
        mlp_w1 + (size_t)l * NGAUSS * HID, mlp_b1 + (size_t)l * HID,
        mlp_w2 + (size_t)l * HID * HID,    mlp_b2 + (size_t)l * HID,
        x, agg);
    // t = ssp(agg @ cf_w2[l] + cf_b2[l])  (t stored in x)
    k_node_gemm<1, true><<<nodeBlocks, 32 * WPB, 0, stream>>>(
        agg, cf_w2 + (size_t)l * HID * HID, cf_b2 + (size_t)l * HID, x, N);
    // h += t @ int_w[l] + int_b[l]
    k_node_gemm<2, true><<<nodeBlocks, 32 * WPB, 0, stream>>>(
        x, int_w + (size_t)l * HID * HID, int_b + (size_t)l * HID, h, N);
  }

  k_zero<<<(G * OUTD + NB - 1) / NB, NB, 0, stream>>>(gsum, G * OUTD);
  k_output<<<G, 32 * OWPB, 0, stream>>>(h, out_w1, out_b1, out_w2, out_b2, gsum);
  k_proj<<<(G * OUTD + NB - 1) / NB, NB, 0, stream>>>(gsum, meta, proj_w, proj_b,
                                                      (float*)d_out, G);
}
